// DCNv2SkipAligner_70420283785814
// MI455X (gfx1250) — matrix operand order repaired and checked by reading, not compiled, not run on hardware
//
#include <hip/hip_runtime.h>
#include <hip/hip_bf16.h>
#include <math.h>

typedef _Float16 v16h __attribute__((ext_vector_type(16)));
typedef _Float16 v8h  __attribute__((ext_vector_type(8)));
typedef float    v8f  __attribute__((ext_vector_type(8)));

#define EPSV 1e-05f
constexpr int Bn = 2, Cn = 64, On = 64, Hn = 192, Wn = 192, Kn = 9;
constexpr int HW = Hn * Wn;

// ---------------------------------------------------------------------------
// Stage 0: NCHW -> NHWC transpose of x (so channel gathers are contiguous)
// ---------------------------------------------------------------------------
__global__ __launch_bounds__(256) void k_transpose(const float* __restrict__ x,
                                                   float* __restrict__ xT) {
  int idx = blockIdx.x * 256 + threadIdx.x;      // over B*H*W*C in NHWC order
  int c = idx & 63;
  int rest = idx >> 6;                           // b*HW + p
  int b = rest / HW;
  int p = rest - b * HW;
  xT[idx] = x[((b * Cn + c) * HW) + p];
}

// ---------------------------------------------------------------------------
// Stage 0b: pack dcn_w (O,C,3,3) fp32 -> f16 laid out [o][k][c]
// so each 8-K run of a WMMA B-fragment (fixed o) is 16B contiguous.
// ---------------------------------------------------------------------------
__global__ __launch_bounds__(256) void k_wpack(const float* __restrict__ dcn_w,
                                               _Float16* __restrict__ wB) {
  int idx = blockIdx.x * 256 + threadIdx.x;      // over O*9*64
  int c = idx & 63;
  int k = (idx >> 6) % 9;
  int o = idx / 576;
  wB[idx] = (_Float16)dcn_w[o * 576 + c * 9 + k];
}

// ---------------------------------------------------------------------------
// Stage 1: fused depthwise 3x3 conv + BN + ReLU + 1x1 conv (64->27) -> om
// ---------------------------------------------------------------------------
__global__ __launch_bounds__(256) void k_offsetmask(
    const float* __restrict__ x, const float* __restrict__ dw_w,
    const float* __restrict__ dw_b, const float* __restrict__ bn_g,
    const float* __restrict__ bn_b, const float* __restrict__ bn_m,
    const float* __restrict__ bn_v, const float* __restrict__ pw_w,
    const float* __restrict__ pw_b, float* __restrict__ om) {
  int pix = blockIdx.x * 256 + threadIdx.x;      // 0 .. B*HW
  int b = pix / HW;
  int p = pix - b * HW;
  int y = p / Wn;
  int xw = p - y * Wn;

  float acc[27];
#pragma unroll
  for (int o = 0; o < 27; ++o) acc[o] = 0.f;

  for (int c = 0; c < Cn; ++c) {
    const float* xc = x + ((long)(b * Cn + c)) * HW;
    float hv = dw_b[c];
#pragma unroll
    for (int ky = 0; ky < 3; ++ky) {
      int yy = y + ky - 1;
      if ((unsigned)yy < (unsigned)Hn) {
        const float* xr = xc + yy * Wn;
#pragma unroll
        for (int kx = 0; kx < 3; ++kx) {
          int xx = xw + kx - 1;
          if ((unsigned)xx < (unsigned)Wn)
            hv += xr[xx] * dw_w[c * 9 + ky * 3 + kx];
        }
      }
    }
    hv = (hv - bn_m[c]) * rsqrtf(bn_v[c] + EPSV) * bn_g[c] + bn_b[c];
    hv = fmaxf(hv, 0.f);
#pragma unroll
    for (int o = 0; o < 27; ++o) acc[o] += hv * pw_w[o * Cn + c];
  }
#pragma unroll
  for (int o = 0; o < 27; ++o)
    om[((long)(b * 27 + o)) * HW + p] = acc[o] + pw_b[o];
}

// ---------------------------------------------------------------------------
// Stage 2: deformable gather + main GEMM via WMMA.
// Block = 128 threads = 4 waves. Each wave: 16 pixels (M-tile) x O=64 (4 N-tiles).
// K-dim = 576 ordered (k outer, c inner); per deform point k: build 16x64 f16
// A-chunk in LDS, then 2 K-steps of v_wmma_f32_16x16x32_f16 per N-tile.
// ---------------------------------------------------------------------------
union AFrag {
  v16h v;
  v8h h[2];
};

__global__ __launch_bounds__(128) void k_main(
    const float* __restrict__ xT, const float* __restrict__ om,
    const _Float16* __restrict__ wB, const float* __restrict__ dcn_b,
    float* __restrict__ out_pre) {
  __shared__ int sOff[4][16][4];
  __shared__ float sWgt[4][16][4];
  __shared__ __align__(16) _Float16 sA[4][16][72];  // 64 + 8 pad (bank-conflict free)

  const int wv = threadIdx.x >> 5;
  const int lane = threadIdx.x & 31;
  const long tilebase = ((long)blockIdx.x * 4 + wv) * 16;  // flat pixel base
  const int b = (int)(tilebase / HW);
  const int pbase = (int)(tilebase - (long)b * HW);        // tile never crosses b

  v8f acc[4] = {};

  for (int k = 0; k < Kn; ++k) {
    // ---- phase A: per-pixel sample params (lanes 0..15) ----
    if (lane < 16) {
      int p = pbase + lane;
      int y = p / Wn;
      int xw = p - y * Wn;
      float dy = om[((long)(b * 27 + 2 * k)) * HW + p];
      float dx = om[((long)(b * 27 + 2 * k + 1)) * HW + p];
      float mr = om[((long)(b * 27 + 18 + k)) * HW + p];
      float m = 1.f / (1.f + __expf(-mr));
      float py = (float)y + (float)(k / 3 - 1) + dy;
      float px = (float)xw + (float)(k % 3 - 1) + dx;
      float fy0 = floorf(py), fx0 = floorf(px);
      int y0 = (int)fy0, x0 = (int)fx0;
      float wy1 = py - fy0, wx1 = px - fx0;
      float wy0 = 1.f - wy1, wx0 = 1.f - wx1;
      int ys[2] = {y0, y0 + 1};
      int xs[2] = {x0, x0 + 1};
      float wy[2] = {wy0, wy1};
      float wx[2] = {wx0, wx1};
#pragma unroll
      for (int jy = 0; jy < 2; ++jy) {
#pragma unroll
        for (int jx = 0; jx < 2; ++jx) {
          int j = jy * 2 + jx;
          bool ok = ((unsigned)ys[jy] < (unsigned)Hn) &&
                    ((unsigned)xs[jx] < (unsigned)Wn);
          sOff[wv][lane][j] = ok ? (((b * Hn + ys[jy]) * Wn + xs[jx]) << 6) : -1;
          sWgt[wv][lane][j] = wy[jy] * wx[jx] * m;
        }
      }
    }
    __syncthreads();

    // ---- phase B: build A-chunk (16 pixels x 64 channels, f16) ----
#pragma unroll
    for (int h2 = 0; h2 < 2; ++h2) {
      int c = lane + 32 * h2;
      for (int p = 0; p < 16; ++p) {
        float v = 0.f;
#pragma unroll
        for (int j = 0; j < 4; ++j) {
          int off = sOff[wv][p][j];
          if (off >= 0) v += sWgt[wv][p][j] * xT[off + c];
        }
        sA[wv][p][c] = (_Float16)v;
      }
    }
    __syncthreads();

    // ---- phase C: 2 K-steps x 4 N-tiles of WMMA ----
    const int half = lane >> 4;
    const int mrow = lane & 15;
#pragma unroll
    for (int ks = 0; ks < 2; ++ks) {
      // ISA 16-bit A layout: elems 0..7 = K(ks*32 + half*8 ..), 8..15 = +16
      AFrag a;
      const _Float16* arow = &sA[wv][mrow][0];
      a.h[0] = *(const v8h*)(arow + ks * 32 + half * 8);
      a.h[1] = *(const v8h*)(arow + ks * 32 + 16 + half * 8);
      int koff = k * 64 + ks * 32 + half * 8;
#pragma unroll
      for (int nt = 0; nt < 4; ++nt) {
        int n = nt * 16 + mrow;  // B layout: lane = N col, K same as A (half*8.., +16)
        v8h b0 = *(const v8h*)(wB + n * 576 + koff);
        v8h b1 = *(const v8h*)(wB + n * 576 + koff + 16);
        v16h bfrag = __builtin_shufflevector(b0, b1, 0, 1, 2, 3, 4, 5, 6, 7, 8, 9, 10, 11, 12, 13, 14, 15);
        acc[nt] = __builtin_amdgcn_wmma_f32_16x16x32_f16(
            false, a.v, false, bfrag, (short)0, acc[nt], false, false);
      }
    }
    __syncthreads();
  }

  // ---- epilogue: add bias, store (8 consecutive pixels per lane per N-tile)
  const int half = lane >> 4;
  const int mrow = lane & 15;
#pragma unroll
  for (int nt = 0; nt < 4; ++nt) {
    int o = nt * 16 + mrow;
    float bias = dcn_b[o];
    v8f r = acc[nt];
#pragma unroll
    for (int i = 0; i < 8; ++i) r[i] += bias;
    float* dst = out_pre + ((long)(b * On + o)) * HW + pbase + half * 8;
    *(v8f*)dst = r;
  }
}

// ---------------------------------------------------------------------------
// Stage 3: per-(b,o) instance norm + ReLU. One block per (b,o).
// ---------------------------------------------------------------------------
__global__ __launch_bounds__(256) void k_instnorm(const float* __restrict__ pre,
                                                  float* __restrict__ out) {
  const int bo = blockIdx.x;
  const float* src = pre + (long)bo * HW;
  float s = 0.f, s2 = 0.f;
  for (int i = threadIdx.x; i < HW; i += 256) {
    float v = src[i];
    s += v;
    s2 += v * v;
  }
  __shared__ float rs[256], rs2[256];
  rs[threadIdx.x] = s;
  rs2[threadIdx.x] = s2;
  __syncthreads();
  for (int st = 128; st > 0; st >>= 1) {
    if (threadIdx.x < st) {
      rs[threadIdx.x] += rs[threadIdx.x + st];
      rs2[threadIdx.x] += rs2[threadIdx.x + st];
    }
    __syncthreads();
  }
  __shared__ float s_mean, s_inv;
  if (threadIdx.x == 0) {
    float mean = rs[0] * (1.f / HW);
    float var = rs2[0] * (1.f / HW) - mean * mean;
    s_mean = mean;
    s_inv = rsqrtf(var + EPSV);
  }
  __syncthreads();
  float mean = s_mean, inv = s_inv;
  float* dst = out + (long)bo * HW;
  for (int i = threadIdx.x; i < HW; i += 256)
    dst[i] = fmaxf((src[i] - mean) * inv, 0.f);
}

// ---------------------------------------------------------------------------
extern "C" void kernel_launch(void* const* d_in, const int* in_sizes, int n_in,
                              void* d_out, int out_size, void* d_ws,
                              size_t ws_size, hipStream_t stream) {
  (void)in_sizes; (void)n_in; (void)out_size; (void)ws_size;
  const float* x     = (const float*)d_in[0];
  const float* dw_w  = (const float*)d_in[1];
  const float* dw_b  = (const float*)d_in[2];
  const float* bn_g  = (const float*)d_in[3];
  const float* bn_b  = (const float*)d_in[4];
  const float* bn_m  = (const float*)d_in[5];
  const float* bn_v  = (const float*)d_in[6];
  const float* pw_w  = (const float*)d_in[7];
  const float* pw_b  = (const float*)d_in[8];
  const float* dcn_w = (const float*)d_in[9];
  const float* dcn_b = (const float*)d_in[10];
  float* out = (float*)d_out;

  char* ws = (char*)d_ws;
  size_t off = 0;
  auto alloc = [&](size_t bytes) {
    char* p = ws + off;
    off += (bytes + 255) & ~(size_t)255;
    return p;
  };
  float*    xT      = (float*)alloc((size_t)Bn * HW * Cn * 4);
  float*    om      = (float*)alloc((size_t)Bn * 27 * HW * 4);
  _Float16* wB      = (_Float16*)alloc((size_t)On * 576 * 2);
  float*    out_pre = (float*)alloc((size_t)Bn * On * HW * 4);

  k_transpose<<<(Bn * HW * Cn) / 256, 256, 0, stream>>>(x, xT);
  k_wpack<<<(On * 576) / 256, 256, 0, stream>>>(dcn_w, wB);
  k_offsetmask<<<(Bn * HW) / 256, 256, 0, stream>>>(x, dw_w, dw_b, bn_g, bn_b,
                                                    bn_m, bn_v, pw_w, pw_b, om);
  k_main<<<(Bn * HW) / 64, 128, 0, stream>>>(xT, om, wB, dcn_b, out_pre);
  k_instnorm<<<Bn * On, 256, 0, stream>>>(out_pre, out);
}
